// glm_MultiHeadAttention1D_10015863734572
// MI455X (gfx1250) — matrix operand order repaired and checked by reading, not compiled, not run on hardware
//
#include <hip/hip_runtime.h>
#include <math.h>

// ---------------------------------------------------------------------------
// GLM MultiHeadAttention block for MI455X (gfx1250, wave32, WMMA bf16)
// S=2048, B=1, H=4096, NH=32, HD=128
// ---------------------------------------------------------------------------

#define S_LEN 2048
#define H_DIM 4096
#define NHEAD 32
#define HD    128
#define QKV_O (3 * H_DIM)   // 12288

typedef __bf16 bf16;
typedef __attribute__((ext_vector_type(8)))  __bf16 v8bf;
typedef __attribute__((ext_vector_type(16))) __bf16 v16bf;
typedef __attribute__((ext_vector_type(8)))  float  v8f;
typedef __attribute__((ext_vector_type(16))) float  v16f;

#if defined(__has_builtin)
#if __has_builtin(__builtin_amdgcn_perm)
#define HAVE_PERM 1
#endif
#endif

// ---- fp32 pair -> packed bf16 dword (round-to-nearest via +0x8000 + perm) --
static __device__ __forceinline__ unsigned pack2bf(float f0, float f1) {
    unsigned u0 = __builtin_bit_cast(unsigned, f0) + 0x8000u;
    unsigned u1 = __builtin_bit_cast(unsigned, f1) + 0x8000u;
#ifdef HAVE_PERM
    // dest bytes {0,1} = u0 bytes {2,3}; dest bytes {2,3} = u1 bytes {2,3}
    return __builtin_amdgcn_perm(u1, u0, 0x07060302u);   // v_perm_b32
#else
    return (u0 >> 16) | (u1 & 0xFFFF0000u);
#endif
}

static __device__ __forceinline__ bf16 f2bf(float f) {
    unsigned u = __builtin_bit_cast(unsigned, f);
    u += 0x7FFFu + ((u >> 16) & 1u);                      // RNE (epilogue only)
    unsigned short h = (unsigned short)(u >> 16);
    return __builtin_bit_cast(bf16, h);
}

static __device__ __forceinline__ v8f vzero8() {
    v8f x;
#pragma unroll
    for (int i = 0; i < 8; ++i) x[i] = 0.0f;
    return x;
}

static __device__ __forceinline__ v8f wmma_bf16(v16bf a, v16bf b, v8f c) {
    // D = A(16x32 bf16) * B(32x16 bf16) + C(16x16 f32)
    return __builtin_amdgcn_wmma_f32_16x16x32_bf16(
        /*neg_a=*/false, a, /*neg_b=*/false, b,
        /*c_mod=*/(short)0, c, /*reuse_a=*/false, /*reuse_b=*/false);
}

// 16 floats -> v16bf with 8 perm-packs (no per-element re-packing)
static __device__ __forceinline__ v16bf pack16(const v16f& f) {
    union { unsigned d[8]; v16bf v; } u;
#pragma unroll
    for (int i = 0; i < 8; ++i) u.d[i] = pack2bf(f[2 * i], f[2 * i + 1]);
    return u.v;
}

// concatenate two v8bf halves into a v16bf without element moves
static __device__ __forceinline__ v16bf cat8(v8bf lo, v8bf hi) {
    return __builtin_shufflevector(lo, hi, 0, 1, 2, 3, 4, 5, 6, 7,
                                   8, 9, 10, 11, 12, 13, 14, 15);
}

// ---------------------------------------------------------------------------
// Kernel A: bulk fp32 -> bf16 tensor convert (8 elems/thread, b128 I/O)
// ---------------------------------------------------------------------------
__global__ __launch_bounds__(256)
void cvt_bf16_kernel(const float* __restrict__ src, bf16* __restrict__ dst, int n8) {
    int i = blockIdx.x * blockDim.x + threadIdx.x;
    if (i >= n8) return;
    v8f f = *(const v8f*)(src + (size_t)i * 8);
    union { unsigned d[4]; v8bf v; } u;
#pragma unroll
    for (int k = 0; k < 4; ++k) u.d[k] = pack2bf(f[2 * k], f[2 * k + 1]);
    *(v8bf*)(dst + (size_t)i * 8) = u.v;
}

// ---------------------------------------------------------------------------
// Kernel 0: rotary cos/sin tables  [S][32];  inv_freq[j] = 10000^(-j/32)
// ---------------------------------------------------------------------------
__global__ void rope_tables_kernel(float* __restrict__ cosT, float* __restrict__ sinT) {
    int idx = blockIdx.x * blockDim.x + threadIdx.x;
    if (idx >= S_LEN * 32) return;
    int p = idx >> 5;
    int j = idx & 31;
    float inv = __expf(-((float)j / 32.0f) * 9.210340371976184f);  // ln(10000)
    float ang = (float)p * inv;
    cosT[idx] = cosf(ang);
    sinT[idx] = sinf(ang);
}

// ---------------------------------------------------------------------------
// Kernel 1: fused QKV GEMM + bias + 2D rotary + q-scale, bf16 outputs.
// A = pre-converted bf16 hidden states; B = fp32 weights perm-packed to bf16.
// Each wave computes a 64x64 tile (4x4 WMMA accumulators).
//   Q -> Qb[head][s][d]  (1/sqrt(HD) folded in, rotary applied)
//   K -> Kb[head][s][d]  (rotary applied)
//   V -> Vt[head][d][s]  (transposed for the P*V WMMA B operand)
// ---------------------------------------------------------------------------
__global__ __launch_bounds__(128)
void qkv_rope_kernel(const bf16* __restrict__ hsb, const float* __restrict__ w,
                     const float* __restrict__ bqkv, const int* __restrict__ pos_ids,
                     const float* __restrict__ cosT, const float* __restrict__ sinT,
                     bf16* __restrict__ Qb, bf16* __restrict__ Kb, bf16* __restrict__ Vt)
{
    const int lane = threadIdx.x & 31;
    const int wave = threadIdx.x >> 5;
    const int NT   = QKV_O / 64;                 // 192 n-tiles
    const int job  = blockIdx.x * 4 + wave;      // 6144 wave jobs
    if (job >= (S_LEN / 64) * NT) return;
    const int mt = job / NT, nb = job % NT;
    const int m0 = mt * 64, n0 = nb * 64;
    const int ln = lane & 15, lh = lane >> 4;
    const int kb8 = lh * 8, kb16 = lh * 16;

    v8f acc[4][4];
#pragma unroll
    for (int ms = 0; ms < 4; ++ms)
#pragma unroll
        for (int t = 0; t < 4; ++t) acc[ms][t] = vzero8();

    for (int k0 = 0; k0 < H_DIM; k0 += 32) {
        v16bf a[4];
#pragma unroll
        for (int ms = 0; ms < 4; ++ms) {
            const bf16* ap = hsb + (size_t)(m0 + ms * 16 + ln) * H_DIM + k0 + kb16;
            a[ms] = cat8(*(const v8bf*)ap, *(const v8bf*)(ap + 8));
        }
#pragma unroll
        for (int t = 0; t < 4; ++t) {
            const float* bp = w + (size_t)(n0 + t * 16 + ln) * H_DIM + k0 + kb16;
            __builtin_prefetch((const char*)bp + 128, 0, 1);   // next k-tile of weights
            v16f bfv = *(const v16f*)bp;
            v16bf b = pack16(bfv);
#pragma unroll
            for (int ms = 0; ms < 4; ++ms)
                acc[ms][t] = wmma_bf16(a[ms], b, acc[ms][t]);
        }
    }

    // ---- epilogue: bias + rotary + write-out -------------------------------
    const int c0    = n0 % 384;          // offset inside one head's [q|k|v] 384 cols
    const int head  = n0 / 384;
    const int sec   = c0 / 128;          // 0=q 1=k 2=v
    const int dbase = c0 % 128;          // 0 or 64 within head dim (64-tile aligned)
    const float qscale = 0.08838834764831845f;   // 1/sqrt(128)

    float bias[4];
#pragma unroll
    for (int t = 0; t < 4; ++t) bias[t] = bqkv[n0 + t * 16 + ln];

    if (sec == 2) {
        // V section: store transposed Vt[head][d][s]
#pragma unroll
        for (int ms = 0; ms < 4; ++ms)
#pragma unroll
            for (int t = 0; t < 4; ++t) {
                const int d = dbase + t * 16 + ln;
                bf16* dst = Vt + ((size_t)head * HD + d) * S_LEN;
#pragma unroll
                for (int e = 0; e < 8; ++e) {
                    const int srow = m0 + ms * 16 + e + 8 * lh;
                    dst[srow] = f2bf(acc[ms][t][e] + bias[t]);
                }
            }
    } else {
        bf16* dst = (sec == 0) ? Qb : Kb;
        const int half = dbase / 64;                 // 0: pos ids, 1: block ids
        const float sc = (sec == 0) ? qscale : 1.0f;
#pragma unroll
        for (int ms = 0; ms < 4; ++ms) {
#pragma unroll
            for (int e = 0; e < 8; ++e) {
                const int srow = m0 + ms * 16 + e + 8 * lh;
                int pid = pos_ids[half * S_LEN + srow] & (S_LEN - 1);
                const float c0v = cosT[pid * 32 + ln],      s0v = sinT[pid * 32 + ln];
                const float c1v = cosT[pid * 32 + 16 + ln], s1v = sinT[pid * 32 + 16 + ln];
                const float x0 = acc[ms][0][e] + bias[0];
                const float x1 = acc[ms][1][e] + bias[1];
                const float x2 = acc[ms][2][e] + bias[2];
                const float x3 = acc[ms][3][e] + bias[3];
                // out[d] = x[d]*cos - x[d+32]*sin ; out[d+32] = x[d+32]*cos + x[d]*sin
                const float r0 = (x0 * c0v - x2 * s0v) * sc;
                const float r2 = (x2 * c0v + x0 * s0v) * sc;
                const float r1 = (x1 * c1v - x3 * s1v) * sc;
                const float r3 = (x3 * c1v + x1 * s1v) * sc;
                bf16* row = dst + ((size_t)head * S_LEN + srow) * HD + dbase;
                row[0 * 16 + ln] = f2bf(r0);
                row[1 * 16 + ln] = f2bf(r1);
                row[2 * 16 + ln] = f2bf(r2);
                row[3 * 16 + ln] = f2bf(r3);
            }
        }
    }
}

// ---------------------------------------------------------------------------
// Kernel 2: causal flash attention. One wave per (head, 16-query tile).
// Key blocks of 32; Q*K^T via 8 WMMAs, online softmax (fp32, xor-shuffle
// row reductions within 16-lane halves), P re-laid C->A via 1KB LDS,
// P*V via 8 WMMAs against transposed V. Output bf16 ctx [S][H].
// ---------------------------------------------------------------------------
__global__ __launch_bounds__(128)
void attn_kernel(const bf16* __restrict__ Qb, const bf16* __restrict__ Kb,
                 const bf16* __restrict__ Vt, bf16* __restrict__ ctxb)
{
    __shared__ bf16 pbuf[4][16][32];             // per-wave P staging
    const int lane = threadIdx.x & 31;
    const int wave = threadIdx.x >> 5;
    const int job  = blockIdx.x * 4 + wave;      // 4096 wave jobs
    if (job >= NHEAD * (S_LEN / 16)) return;
    const int head = job >> 7;                   // /128
    const int qt   = job & 127;
    const int m0   = qt * 16;
    const int ln = lane & 15, lh = lane >> 4;
    const int kb8 = lh * 8, kb16 = lh * 16;

    // Q tile 16x128 in A layout (4 chunks of K=32)
    const bf16* qbase = Qb + ((size_t)head * S_LEN + m0 + ln) * HD;
    v16bf q[4];
#pragma unroll
    for (int c = 0; c < 4; ++c)
        q[c] = cat8(*(const v8bf*)(qbase + c * 32 + kb16),
                    *(const v8bf*)(qbase + c * 32 + kb16 + 8));

    float rmax[8], rsum[8];
    int   rowi[8];
    v8f   o[8];
#pragma unroll
    for (int e = 0; e < 8; ++e) { rmax[e] = -1.0e30f; rsum[e] = 0.0f; rowi[e] = m0 + e + 8 * lh; }
#pragma unroll
    for (int t = 0; t < 8; ++t) o[t] = vzero8();

    const int nkb = (m0 + 15) / 32 + 1;          // causal: keys <= m0+15
    for (int kb = 0; kb < nkb; ++kb) {
        // ---- scores S = Q K^T, 16x32 tile --------------------------------
        v8f s0 = vzero8(), s1 = vzero8();
        const bf16* k0p = Kb + ((size_t)head * S_LEN + kb * 32 + ln) * HD + kb16;
        const bf16* k1p = k0p + (size_t)16 * HD;
#pragma unroll
        for (int c = 0; c < 4; ++c) {
            v16bf bk0 = *(const v16bf*)(k0p + c * 32);
            s0 = wmma_bf16(q[c], bk0, s0);
            v16bf bk1 = *(const v16bf*)(k1p + c * 32);
            s1 = wmma_bf16(q[c], bk1, s1);
        }

        // ---- online softmax (scale already folded into Q) ----------------
        const int j0 = kb * 32 + ln, j1 = j0 + 16;
        float p0[8], p1[8], alpha[8];
#pragma unroll
        for (int e = 0; e < 8; ++e) {
            float a0 = (j0 <= rowi[e]) ? s0[e] : -3.0e38f;
            float a1 = (j1 <= rowi[e]) ? s1[e] : -3.0e38f;
            float m = fmaxf(a0, a1);
            m = fmaxf(m, __shfl_xor(m, 1));
            m = fmaxf(m, __shfl_xor(m, 2));
            m = fmaxf(m, __shfl_xor(m, 4));
            m = fmaxf(m, __shfl_xor(m, 8));      // row max over 16-lane half
            const float nm = fmaxf(rmax[e], m);
            alpha[e] = __expf(rmax[e] - nm);
            rmax[e]  = nm;
            p0[e] = (j0 <= rowi[e]) ? __expf(s0[e] - nm) : 0.0f;
            p1[e] = (j1 <= rowi[e]) ? __expf(s1[e] - nm) : 0.0f;
            float r = p0[e] + p1[e];
            r += __shfl_xor(r, 1);
            r += __shfl_xor(r, 2);
            r += __shfl_xor(r, 4);
            r += __shfl_xor(r, 8);
            rsum[e] = rsum[e] * alpha[e] + r;
        }
#pragma unroll
        for (int t = 0; t < 8; ++t)
#pragma unroll
            for (int e = 0; e < 8; ++e) o[t][e] *= alpha[e];

        // ---- P: C layout -> LDS -> A layout (wave-local, DS ops in-order) -
#pragma unroll
        for (int e = 0; e < 8; ++e) {
            unsigned pp = pack2bf(p0[e], p1[e]);
            pbuf[wave][e + 8 * lh][ln]      = __builtin_bit_cast(bf16, (unsigned short)(pp & 0xFFFFu));
            pbuf[wave][e + 8 * lh][16 + ln] = __builtin_bit_cast(bf16, (unsigned short)(pp >> 16));
        }
        v16bf pa;
        {
            const bf16* pr = &pbuf[wave][ln][0];
            pa = cat8(*(const v8bf*)(pr + kb16), *(const v8bf*)(pr + kb16 + 8));
        }

        // ---- O += P V  (V transposed: contiguous 32B loads per lane) ------
        const bf16* vb0 = Vt + (size_t)head * HD * S_LEN + (size_t)ln * S_LEN + kb * 32 + kb16;
#pragma unroll
        for (int t = 0; t < 8; ++t) {
            v16bf vb = *(const v16bf*)(vb0 + (size_t)t * 16 * S_LEN);
            o[t] = wmma_bf16(pa, vb, o[t]);
        }
    }

    // ---- normalize + store bf16 context [S][H] ---------------------------
#pragma unroll
    for (int e = 0; e < 8; ++e) rsum[e] = 1.0f / rsum[e];
#pragma unroll
    for (int t = 0; t < 8; ++t) {
        const int col = head * HD + t * 16 + ln;
#pragma unroll
        for (int e = 0; e < 8; ++e) {
            const int srow = m0 + e + 8 * lh;
            ctxb[(size_t)srow * H_DIM + col] = f2bf(o[t][e] * rsum[e]);
        }
    }
}

// ---------------------------------------------------------------------------
// Kernel 3: dense projection  out[s,o] = ctx[s,:] . w_dense[o,:] + b_dense[o]
// 64x64 tile per wave, bf16 A (context), fp32 weights perm-packed to bf16.
// ---------------------------------------------------------------------------
__global__ __launch_bounds__(128)
void dense_kernel(const bf16* __restrict__ ctxb, const float* __restrict__ w,
                  const float* __restrict__ bias, float* __restrict__ out)
{
    const int lane = threadIdx.x & 31;
    const int wave = threadIdx.x >> 5;
    const int NT   = H_DIM / 64;                 // 64
    const int job  = blockIdx.x * 4 + wave;      // 2048 wave jobs
    if (job >= (S_LEN / 64) * NT) return;
    const int mt = job / NT, nb = job % NT;
    const int m0 = mt * 64, n0 = nb * 64;
    const int ln = lane & 15, lh = lane >> 4;
    const int kb8 = lh * 8, kb16 = lh * 16;

    v8f acc[4][4];
#pragma unroll
    for (int ms = 0; ms < 4; ++ms)
#pragma unroll
        for (int t = 0; t < 4; ++t) acc[ms][t] = vzero8();

    for (int k0 = 0; k0 < H_DIM; k0 += 32) {
        v16bf a[4];
#pragma unroll
        for (int ms = 0; ms < 4; ++ms) {
            const bf16* ap = ctxb + (size_t)(m0 + ms * 16 + ln) * H_DIM + k0 + kb16;
            a[ms] = cat8(*(const v8bf*)ap, *(const v8bf*)(ap + 8));
        }
#pragma unroll
        for (int t = 0; t < 4; ++t) {
            const float* bp = w + (size_t)(n0 + t * 16 + ln) * H_DIM + k0 + kb16;
            __builtin_prefetch((const char*)bp + 128, 0, 1);
            v16f bfv = *(const v16f*)bp;
            v16bf b = pack16(bfv);
#pragma unroll
            for (int ms = 0; ms < 4; ++ms)
                acc[ms][t] = wmma_bf16(a[ms], b, acc[ms][t]);
        }
    }

#pragma unroll
    for (int t = 0; t < 4; ++t) {
        const int col = n0 + t * 16 + ln;
        const float bt = bias[col];
#pragma unroll
        for (int ms = 0; ms < 4; ++ms)
#pragma unroll
            for (int e = 0; e < 8; ++e) {
                const int srow = m0 + ms * 16 + e + 8 * lh;
                out[(size_t)srow * H_DIM + col] = acc[ms][t][e] + bt;
            }
    }
}

// ---------------------------------------------------------------------------
// Host launcher
// inputs: 0 hidden_states[S,1,H] f32 | 1 position_ids[1,2,S] int |
//         2 attention_mask (unused: causal derived analytically) |
//         3 layer_id (unused: coeff cancels in softmax) |
//         4 w_qkv[3H,H] f32 | 5 b_qkv[3H] f32 | 6 w_dense[H,H] f32 | 7 b_dense[H] f32
// output: [S,1,H] f32
// ---------------------------------------------------------------------------
extern "C" void kernel_launch(void* const* d_in, const int* in_sizes, int n_in,
                              void* d_out, int out_size, void* d_ws, size_t ws_size,
                              hipStream_t stream)
{
    const float* hs      = (const float*)d_in[0];
    const int*   pos_ids = (const int*)d_in[1];
    const float* w_qkv   = (const float*)d_in[4];
    const float* b_qkv   = (const float*)d_in[5];
    const float* w_dense = (const float*)d_in[6];
    const float* b_dense = (const float*)d_in[7];
    float* out = (float*)d_out;

    char* ws = (char*)d_ws;
    float* cosT = (float*)ws;  ws += (size_t)S_LEN * 32 * sizeof(float);
    float* sinT = (float*)ws;  ws += (size_t)S_LEN * 32 * sizeof(float);
    bf16* hsb  = (bf16*)ws;    ws += (size_t)S_LEN * H_DIM * sizeof(bf16);
    bf16* Qb   = (bf16*)ws;    ws += (size_t)NHEAD * S_LEN * HD * sizeof(bf16);
    bf16* Kb   = (bf16*)ws;    ws += (size_t)NHEAD * S_LEN * HD * sizeof(bf16);
    bf16* Vt   = (bf16*)ws;    ws += (size_t)NHEAD * HD * S_LEN * sizeof(bf16);
    bf16* ctxb = (bf16*)ws;    ws += (size_t)S_LEN * H_DIM * sizeof(bf16);

    rope_tables_kernel<<<dim3((S_LEN * 32 + 255) / 256), dim3(256), 0, stream>>>(cosT, sinT);

    const int n8 = (S_LEN * H_DIM) / 8;                        // hidden -> bf16 once
    cvt_bf16_kernel<<<dim3((n8 + 255) / 256), dim3(256), 0, stream>>>(hs, hsb, n8);

    const int qkv_jobs = (S_LEN / 64) * (QKV_O / 64);          // 6144 waves
    qkv_rope_kernel<<<dim3(qkv_jobs / 4), dim3(128), 0, stream>>>(
        hsb, w_qkv, b_qkv, pos_ids, cosT, sinT, Qb, Kb, Vt);

    const int attn_jobs = NHEAD * (S_LEN / 16);                // 4096 waves
    attn_kernel<<<dim3(attn_jobs / 4), dim3(128), 0, stream>>>(Qb, Kb, Vt, ctxb);

    const int dense_jobs = (S_LEN / 64) * (H_DIM / 64);        // 2048 waves
    dense_kernel<<<dim3(dense_jobs / 4), dim3(128), 0, stream>>>(ctxb, w_dense, b_dense, out);
}
